// GraphSAGE_71906342469641
// MI455X (gfx1250) — compile-verified, matrix operand order checked by reading, not run on hardware
//
#include <hip/hip_runtime.h>
#include <hip/hip_bf16.h>

typedef __bf16 v16bf __attribute__((ext_vector_type(16)));
typedef float  v8f   __attribute__((ext_vector_type(8)));

#define NNODES 50000

// ---------------------------------------------------------------------------
// Pack a row-major f32 weight [Kdim x Nout] into WMMA B-fragment layout, bf16.
// Tile (kc, nt) covers K in [kc*32, kc*32+32), N in [nt*16, nt*16+16).
// Per 512-element tile: lane l, elem i ->
//   n = nt*16 + (l&15)
//   k = kc*32 + (i<8 ? (l>>4)*8 + i : 16 + (l>>4)*8 + (i-8))
// Each lane's 16 values are contiguous 32 bytes at
//   Bp + (kc*numNT + nt)*1024 + lane*32   (bytes)
// ---------------------------------------------------------------------------
__global__ void pack_w_kernel(const float* __restrict__ W, int Kdim, int Nout,
                              __bf16* __restrict__ dst) {
    int i = blockIdx.x * blockDim.x + threadIdx.x;
    int total = Kdim * Nout;
    if (i >= total) return;
    int elem = i & 15;
    int lane = (i >> 4) & 31;
    int tile = i >> 9;
    int numNT = Nout >> 4;
    int kc = tile / numNT;
    int nt = tile % numNT;
    int hi = lane >> 4;
    int koff = (elem < 8) ? (hi * 8 + elem) : (16 + hi * 8 + (elem - 8));
    int k = kc * 32 + koff;
    int n = nt * 16 + (lane & 15);
    dst[i] = (__bf16)W[(size_t)k * Nout + n];
}

// Convert features f32 [N,128] -> bf16 into hcat1[:, 0:128] (row stride 256).
__global__ void cvt_feat_kernel(const float* __restrict__ src,
                                __bf16* __restrict__ hcat1) {
    int i = blockIdx.x * blockDim.x + threadIdx.x;
    if (i >= NNODES * 128) return;
    int node = i >> 7;
    int col  = i & 127;
    hcat1[(size_t)node * 256 + col] = (__bf16)src[i];
}

// A fragment: 16-bit A 16x32 layout.
// lanes 0-15 (hi=0): K 0-7 in elems 0-7, K 16-23 in elems 8-15
// lanes 16-31(hi=1): K 8-15 in elems 0-7, K 24-31 in elems 8-15
// abase = row + hi*8 elements; per k-chunk kc, halves at byte offsets
// kc*64 and kc*64+32 -> compile-time immediates once unrolled.
__device__ __forceinline__ v16bf load_a_frag(const char* __restrict__ abase,
                                             int byteOff) {
    union { v16bf v; uint4 q[2]; } u;
    u.q[0] = *(const uint4*)(abase + byteOff);
    u.q[1] = *(const uint4*)(abase + byteOff + 32);
    return u.v;
}

__device__ __forceinline__ v8f wmma_bf16(v16bf a, v16bf b, v8f c) {
    return __builtin_amdgcn_wmma_f32_16x16x32_bf16(
        false, a, false, b, (short)0, c, false, false);
}

// ---------------------------------------------------------------------------
// Aggregator: one wave per (node, 64-col tile).
// M-tile rows = the node's 16 sampled neighbors (gathered rows of hsrc).
// C[16 x 64] = A[16 x KDIM] @ Wp[KDIM x NOUT] (bf16 x bf16 -> f32).
// Epilogue: pooled[c] = relu(max_rows(C[:,c]) + bias[c]) -> bf16 dst row.
// (max_k relu(x+b) == relu(max_k x + b): relu monotone, b uniform over k.)
// ---------------------------------------------------------------------------
template <int KDIM, int NOUT>
__global__ void sage_agg_kernel(const __bf16* __restrict__ hsrc, int srcStride,
                                const int* __restrict__ nbr,
                                const __bf16* __restrict__ Bp,
                                const float* __restrict__ bias,
                                __bf16* __restrict__ dst, int dstStride,
                                int dstColOff) {
    constexpr int numCT = NOUT / 64;
    constexpr int numNT = NOUT / 16;
    constexpr int numKC = KDIM / 32;
    int wid  = (blockIdx.x * blockDim.x + threadIdx.x) >> 5;
    int lane = threadIdx.x & 31;
    if (wid >= NNODES * numCT) return;
    int node = wid / numCT;
    int ct   = wid % numCT;
    int m  = lane & 15;
    int hi = lane >> 4;

    const char* abase = (const char*)(hsrc +
        (size_t)nbr[node * 16 + m] * srcStride + hi * 8);
    const char* bbase = (const char*)Bp + (size_t)(ct * 4) * 1024 +
        (size_t)lane * 32;

    v8f z = {0.f, 0.f, 0.f, 0.f, 0.f, 0.f, 0.f, 0.f};
    v8f acc[4] = {z, z, z, z};

#pragma unroll
    for (int kc = 0; kc < numKC; ++kc) {
        v16bf a = load_a_frag(abase, kc * 64);
#pragma unroll
        for (int t = 0; t < 4; ++t) {
            v16bf b = *(const v16bf*)(bbase + (kc * numNT + t) * 1024);
            acc[t] = wmma_bf16(a, b, acc[t]);
        }
    }

#pragma unroll
    for (int t = 0; t < 4; ++t) {
        float mv = acc[t][0];
#pragma unroll
        for (int v = 1; v < 8; ++v) mv = fmaxf(mv, acc[t][v]);
        mv = fmaxf(mv, __shfl_xor(mv, 16, 32));   // combine M 0-7 with 8-15
        if (lane < 16) {
            int c = ct * 64 + t * 16 + lane;
            float r = fmaxf(mv + bias[c], 0.f);
            dst[(size_t)node * dstStride + dstColOff + c] = (__bf16)r;
        }
    }
}

// ---------------------------------------------------------------------------
// Dense linear: one wave per (16-row tile, 64-col tile).
// A rows are contiguous hcat rows (concat(self, pooled) already materialized).
// ---------------------------------------------------------------------------
template <int KDIM, int NOUT, bool RELU, bool OUT_BF16>
__global__ void sage_gemm_kernel(const __bf16* __restrict__ A, int aStride,
                                 const __bf16* __restrict__ Bp,
                                 const float* __restrict__ bias,
                                 void* __restrict__ dst, int dstStride) {
    constexpr int numCT = NOUT / 64;
    constexpr int numNT = NOUT / 16;
    constexpr int numKC = KDIM / 32;
    constexpr int nRowTiles = NNODES / 16;     // 3125 exactly
    int wid  = (blockIdx.x * blockDim.x + threadIdx.x) >> 5;
    int lane = threadIdx.x & 31;
    if (wid >= nRowTiles * numCT) return;
    int rt = wid / numCT;
    int ct = wid % numCT;
    int hi = lane >> 4;

    const char* abase = (const char*)(A +
        (size_t)(rt * 16 + (lane & 15)) * aStride + hi * 8);
    const char* bbase = (const char*)Bp + (size_t)(ct * 4) * 1024 +
        (size_t)lane * 32;

    v8f z = {0.f, 0.f, 0.f, 0.f, 0.f, 0.f, 0.f, 0.f};
    v8f acc[4] = {z, z, z, z};

#pragma unroll
    for (int kc = 0; kc < numKC; ++kc) {
        v16bf a = load_a_frag(abase, kc * 64);
#pragma unroll
        for (int t = 0; t < 4; ++t) {
            v16bf b = *(const v16bf*)(bbase + (kc * numNT + t) * 1024);
            acc[t] = wmma_bf16(a, b, acc[t]);
        }
    }

    // C layout: lane<16 holds M=v (v=0..7) of column (lane&15); lane>=16 M=8+v.
#pragma unroll
    for (int t = 0; t < 4; ++t) {
        int c = ct * 64 + t * 16 + (lane & 15);
        float bv = bias[c];
#pragma unroll
        for (int v = 0; v < 8; ++v) {
            int r = rt * 16 + v + 8 * hi;
            float val = acc[t][v] + bv;
            if (RELU) val = fmaxf(val, 0.f);
            if (OUT_BF16)
                ((__bf16*)dst)[(size_t)r * dstStride + c] = (__bf16)val;
            else
                ((float*)dst)[(size_t)r * dstStride + c] = val;
        }
    }
}

extern "C" void kernel_launch(void* const* d_in, const int* in_sizes, int n_in,
                              void* d_out, int out_size, void* d_ws, size_t ws_size,
                              hipStream_t stream) {
    const float* features = (const float*)d_in[0];
    const int*   nbr      = (const int*)d_in[1];
    const float* Wp1 = (const float*)d_in[2];
    const float* bp1 = (const float*)d_in[3];
    const float* W1  = (const float*)d_in[4];
    const float* b1  = (const float*)d_in[5];
    const float* Wp2 = (const float*)d_in[6];
    const float* bp2 = (const float*)d_in[7];
    const float* W2  = (const float*)d_in[8];
    const float* b2  = (const float*)d_in[9];
    float* out = (float*)d_out;

    char* ws = (char*)d_ws;
    __bf16* hcat1 = (__bf16*)ws;                              // [N,256] bf16: feat|pool1
    __bf16* hcat2 = (__bf16*)(ws + (size_t)25600000);         // [N,512] bf16: h1|pool2
    __bf16* pWp1  = (__bf16*)(ws + (size_t)76800000);         // 128x128
    __bf16* pW1   = pWp1 + 128 * 128;                         // 256x256
    __bf16* pWp2  = pW1 + 256 * 256;                          // 256x256
    __bf16* pW2   = pWp2 + 256 * 256;                         // 512x128

    // --- pack weights into WMMA B-fragment layout (bf16) ---
    pack_w_kernel<<<(128 * 128 + 255) / 256, 256, 0, stream>>>(Wp1, 128, 128, pWp1);
    pack_w_kernel<<<(256 * 256 + 255) / 256, 256, 0, stream>>>(W1, 256, 256, pW1);
    pack_w_kernel<<<(256 * 256 + 255) / 256, 256, 0, stream>>>(Wp2, 256, 256, pWp2);
    pack_w_kernel<<<(512 * 128 + 255) / 256, 256, 0, stream>>>(W2, 512, 128, pW2);

    // --- features -> bf16 in hcat1[:, 0:128] ---
    cvt_feat_kernel<<<(NNODES * 128 + 255) / 256, 256, 0, stream>>>(features, hcat1);

    // --- layer 1 aggregator: pooled1 -> hcat1[:, 128:256] ---
    {
        int units = NNODES * (128 / 64);                 // 100000 waves
        sage_agg_kernel<128, 128><<<(units + 7) / 8, 256, 0, stream>>>(
            hcat1, 256, nbr, pWp1, bp1, hcat1, 256, 128);
    }
    // --- layer 1 linear + relu: h1 -> hcat2[:, 0:256] ---
    {
        int units = (NNODES / 16) * (256 / 64);          // 12500 waves
        sage_gemm_kernel<256, 256, true, true><<<(units + 7) / 8, 256, 0, stream>>>(
            hcat1, 256, pW1, b1, (void*)hcat2, 512);
    }
    // --- layer 2 aggregator: pooled2 -> hcat2[:, 256:512] ---
    {
        int units = NNODES * (256 / 64);                 // 200000 waves
        sage_agg_kernel<256, 256><<<(units + 7) / 8, 256, 0, stream>>>(
            hcat2, 512, nbr, pWp2, bp2, hcat2, 512, 256);
    }
    // --- layer 2 linear (no relu), f32 out ---
    {
        int units = (NNODES / 16) * (128 / 64);          // 6250 waves
        sage_gemm_kernel<512, 128, false, false><<<(units + 7) / 8, 256, 0, stream>>>(
            hcat2, 512, pW2, b2, (void*)out, 128);
    }
}
